// MHCausalSelfAttention_4234837754197
// MI455X (gfx1250) — matrix operand order repaired and checked by reading, not compiled, not run on hardware
//
#include <hip/hip_runtime.h>

// ---------------------------------------------------------------------------
// MH causal self-attention for MI455X (gfx1250, wave32, WMMA bf16)
// N=2, L=2048, C=1024, H=16, D=64
// ---------------------------------------------------------------------------

typedef __attribute__((ext_vector_type(16))) __bf16 v16bf;
typedef __attribute__((ext_vector_type(8)))  __bf16 v8bf;
typedef __attribute__((ext_vector_type(4)))  __bf16 v4bf;
typedef __attribute__((ext_vector_type(8)))  float  v8f;

#define NB  2
#define LL  2048
#define CC  1024
#define HH  16
#define DD  64

// ---- WMMA helper -----------------------------------------------------------
__device__ __forceinline__ v8f wmma_bf16(v16bf a, v16bf b, v8f c) {
  // D = A(16x32 bf16) x B(32x16 bf16) + C(16x16 f32)
  return __builtin_amdgcn_wmma_f32_16x16x32_bf16(
      /*neg_a=*/false, a, /*neg_b=*/false, b,
      /*c_mod=*/(short)0, c, /*reuse_a=*/false, /*reuse_b=*/false);
}

// A-fragment (16x32 bf16) from a row-major row pointer (stride = K elems).
// lane m = lane&15 ; lanes<16 hold K {0..15}, lanes>=16 hold {16..31}
__device__ __forceinline__ v16bf load_a16(const __bf16* rowk, int kA0) {
  v8bf lo = *(const v8bf*)(rowk + kA0);
  v8bf hi = *(const v8bf*)(rowk + 8 + kA0);
  v16bf r;
#pragma unroll
  for (int i = 0; i < 8; ++i) { r[i] = lo[i]; r[i + 8] = hi[i]; }
  return r;
}

// Row reductions across the 16 lanes of a half-wave (rows live per-half).
__device__ __forceinline__ float rmax16(float x) {
#pragma unroll
  for (int m = 1; m <= 8; m <<= 1) x = fmaxf(x, __shfl_xor(x, m, 32));
  return x;
}
__device__ __forceinline__ float rsum16(float x) {
#pragma unroll
  for (int m = 1; m <= 8; m <<= 1) x += __shfl_xor(x, m, 32);
  return x;
}

// ---- Kernel 0: f32 -> bf16 (vector x4) ------------------------------------
__global__ void cvt_bf16x4(const float* __restrict__ in,
                           __bf16* __restrict__ out, int n4) {
  int i = blockIdx.x * blockDim.x + threadIdx.x;
  if (i >= n4) return;
  float4 v = ((const float4*)in)[i];
  v4bf o;
  o[0] = (__bf16)v.x; o[1] = (__bf16)v.y;
  o[2] = (__bf16)v.z; o[3] = (__bf16)v.w;
  ((v4bf*)out)[i] = o;
}

// ---- Kernel 1: QKV GEMM ----------------------------------------------------
// qkv[m][o] = sum_c x[m][c] * w_qkv[o][c],  M = N*L = 4096, O = 3C = 3072
// Each wave: 16(M) x 64(O) tile; block = 8 waves -> 128 x 64.
// grid = (48, 32). Output scattered into Qb/Kb [nh][l][d] and Vt [nh][d][l].
__global__ void qkv_gemm(const __bf16* __restrict__ xb,
                         const __bf16* __restrict__ wb,
                         __bf16* __restrict__ Qb,
                         __bf16* __restrict__ Kb,
                         __bf16* __restrict__ Vt) {
  const int lane = threadIdx.x & 31;
  const int wv   = threadIdx.x >> 5;
  const int hl   = lane >> 4;
  const int mbase = blockIdx.y * 128 + wv * 16;   // row-tile base in M
  const int ob    = blockIdx.x * 64;              // output-channel base
  const int kA0 = hl ? 16 : 0;
  const int kB0 = hl ? 16 : 0;

  const __bf16* arow = xb + (size_t)(mbase + (lane & 15)) * CC;

  v8f acc[4] = {};
#pragma unroll 2
  for (int kk = 0; kk < CC; kk += 32) {
    v16bf a = load_a16(arow + kk, kA0);
#pragma unroll
    for (int f = 0; f < 4; ++f) {
      const __bf16* brow = wb + (size_t)(ob + f * 16 + (lane & 15)) * CC;
      v16bf b = *(const v16bf*)(brow + kk + kB0);
      acc[f] = wmma_bf16(a, b, acc[f]);
    }
  }

  const int s    = ob >> 10;             // 0:Q 1:K 2:V
  const int h    = (ob >> 6) & 15;
  const int n    = mbase >> 11;
  const int lrow = mbase & (LL - 1);

  if (s < 2) {
    __bf16* dst = (s == 0 ? Qb : Kb) + (size_t)(n * HH + h) * (LL * DD);
#pragma unroll
    for (int f = 0; f < 4; ++f) {
      int d = f * 16 + (lane & 15);
#pragma unroll
      for (int v = 0; v < 8; ++v) {
        int row = lrow + v + 8 * hl;
        dst[(size_t)row * DD + d] = (__bf16)acc[f][v];
      }
    }
  } else {
    // V transposed: Vt[nh][d][l] -> lane holds 8 consecutive l positions
    __bf16* dst = Vt + (size_t)(n * HH + h) * (DD * LL);
#pragma unroll
    for (int f = 0; f < 4; ++f) {
      int d = f * 16 + (lane & 15);
      v8bf pk;
#pragma unroll
      for (int v = 0; v < 8; ++v) pk[v] = (__bf16)acc[f][v];
      *(v8bf*)(dst + (size_t)d * LL + lrow + 8 * hl) = pk;
    }
  }
}

// ---- Kernel 2: causal flash attention -------------------------------------
// One wave per 16-row Q tile; block = 4 waves (64 rows); grid = (32, N*H).
__global__ void flash_attn(const __bf16* __restrict__ Qb,
                           const __bf16* __restrict__ Kb,
                           const __bf16* __restrict__ Vt,
                           __bf16* __restrict__ Yb) {
  const int lane = threadIdx.x & 31;
  const int wv   = threadIdx.x >> 5;
  const int hl   = lane >> 4;
  const int nh   = blockIdx.y;
  const int n    = nh >> 4;
  const int h    = nh & 15;
  const int rbase = (blockIdx.x * 4 + wv) * 16;   // Q row-tile base

  const __bf16* Qh = Qb + (size_t)nh * (LL * DD);
  const __bf16* Kh = Kb + (size_t)nh * (LL * DD);
  const __bf16* Vh = Vt + (size_t)nh * (DD * LL);

  __shared__ __bf16 Pld[4][16][32];               // 1 KB per wave

  const int kA0 = hl ? 16 : 0;
  const int kB0 = hl ? 16 : 0;

  // Q fragments (16x64 = two K=32 A-fragments), loaded once
  const __bf16* qrow = Qh + (size_t)(rbase + (lane & 15)) * DD;
  v16bf aQ0 = load_a16(qrow,      kA0);
  v16bf aQ1 = load_a16(qrow + 32, kA0);

  v8f accO[4] = {};
  float m_i[8], l_i[8];
#pragma unroll
  for (int v = 0; v < 8; ++v) { m_i[v] = -1e30f; l_i[v] = 0.f; }

  for (int jb = 0; jb <= rbase + 15; jb += 32) {
    // ---- S = Q K^T for two 16-col subtiles -------------------------------
    float Pv[2][8];
#pragma unroll
    for (int sub = 0; sub < 2; ++sub) {
      const __bf16* krow = Kh + (size_t)(jb + sub * 16 + (lane & 15)) * DD;
      v8f s = {};
      s = wmma_bf16(aQ0, *(const v16bf*)(krow + kB0),      s);
      s = wmma_bf16(aQ1, *(const v16bf*)(krow + 32 + kB0), s);
      int col = jb + sub * 16 + (lane & 15);
#pragma unroll
      for (int v = 0; v < 8; ++v) {
        int row = rbase + v + 8 * hl;
        float sv = s[v] * 0.125f;                 // 1/sqrt(64)
        Pv[sub][v] = (col > row) ? -1e30f : sv;   // causal mask
      }
    }
    // ---- online softmax ---------------------------------------------------
#pragma unroll
    for (int v = 0; v < 8; ++v) {
      float t  = rmax16(fmaxf(Pv[0][v], Pv[1][v]));
      float mn = fmaxf(m_i[v], t);
      float alpha = __expf(m_i[v] - mn);
      m_i[v] = mn;
      float p0 = __expf(Pv[0][v] - mn);
      float p1 = __expf(Pv[1][v] - mn);
      Pv[0][v] = p0; Pv[1][v] = p1;
      l_i[v] = l_i[v] * alpha + rsum16(p0 + p1);
#pragma unroll
      for (int f = 0; f < 4; ++f) accO[f][v] *= alpha;
    }
    // ---- P (C-layout) -> LDS -> A-fragment layout ------------------------
#pragma unroll
    for (int sub = 0; sub < 2; ++sub)
#pragma unroll
      for (int v = 0; v < 8; ++v)
        Pld[wv][v + 8 * hl][sub * 16 + (lane & 15)] = (__bf16)Pv[sub][v];
    // same-wave DS ops are in-order: no barrier needed
    v16bf aP = load_a16(&Pld[wv][lane & 15][0], kA0);
    // ---- O += P V  (B-fragments contiguous thanks to transposed V) -------
#pragma unroll
    for (int f = 0; f < 4; ++f) {
      const __bf16* vrow = Vh + (size_t)(f * 16 + (lane & 15)) * LL;
      accO[f] = wmma_bf16(aP, *(const v16bf*)(vrow + jb + kB0), accO[f]);
    }
  }

  // ---- normalize and store Y [N][L][C], c = h*D + d -----------------------
#pragma unroll
  for (int f = 0; f < 4; ++f) {
    int d = f * 16 + (lane & 15);
#pragma unroll
    for (int v = 0; v < 8; ++v) {
      int row = rbase + v + 8 * hl;
      float y = accO[f][v] / l_i[v];
      Yb[(size_t)(n * LL + row) * CC + h * DD + d] = (__bf16)y;
    }
  }
}

// ---- Kernel 3: output projection (f32 out) --------------------------------
// y[m][o] = sum_c Y[m][c] * w_out[o][c],  grid = (16, 32), block = 256.
__global__ void out_gemm(const __bf16* __restrict__ Yb,
                         const __bf16* __restrict__ wb,
                         float* __restrict__ out) {
  const int lane = threadIdx.x & 31;
  const int wv   = threadIdx.x >> 5;
  const int hl   = lane >> 4;
  const int mbase = blockIdx.y * 128 + wv * 16;
  const int ob    = blockIdx.x * 64;
  const int kA0 = hl ? 16 : 0;
  const int kB0 = hl ? 16 : 0;

  const __bf16* arow = Yb + (size_t)(mbase + (lane & 15)) * CC;

  v8f acc[4] = {};
#pragma unroll 2
  for (int kk = 0; kk < CC; kk += 32) {
    v16bf a = load_a16(arow + kk, kA0);
#pragma unroll
    for (int f = 0; f < 4; ++f) {
      const __bf16* brow = wb + (size_t)(ob + f * 16 + (lane & 15)) * CC;
      acc[f] = wmma_bf16(a, *(const v16bf*)(brow + kk + kB0), acc[f]);
    }
  }
#pragma unroll
  for (int f = 0; f < 4; ++f) {
    int o = ob + f * 16 + (lane & 15);
#pragma unroll
    for (int v = 0; v < 8; ++v) {
      int row = mbase + v + 8 * hl;
      out[(size_t)row * CC + o] = acc[f][v];
    }
  }
}

// ---------------------------------------------------------------------------
extern "C" void kernel_launch(void* const* d_in, const int* in_sizes, int n_in,
                              void* d_out, int out_size, void* d_ws, size_t ws_size,
                              hipStream_t stream) {
  const float* x     = (const float*)d_in[0];   // [N, L, C]
  const float* w_qkv = (const float*)d_in[1];   // [3C, C]
  const float* w_out = (const float*)d_in[2];   // [C, C]
  float* out = (float*)d_out;                   // [N, L, C]

  // workspace layout (bf16 elements), total 24M elems = 48 MB
  __bf16* xb    = (__bf16*)d_ws;                  // N*L*C   = 4M
  __bf16* wqkvb = xb    + (size_t)NB * LL * CC;   // 3C*C    = 3M
  __bf16* woutb = wqkvb + (size_t)3 * CC * CC;    // C*C     = 1M
  __bf16* Qb    = woutb + (size_t)CC * CC;        // N*H*L*D = 4M
  __bf16* Kb    = Qb    + (size_t)NB * HH * LL * DD;
  __bf16* Vt    = Kb    + (size_t)NB * HH * LL * DD;
  __bf16* Yb    = Vt    + (size_t)NB * HH * LL * DD;

  // 0) convert inputs to bf16
  {
    int n4;
    n4 = (NB * LL * CC) / 4;
    cvt_bf16x4<<<(n4 + 255) / 256, 256, 0, stream>>>(x, xb, n4);
    n4 = (3 * CC * CC) / 4;
    cvt_bf16x4<<<(n4 + 255) / 256, 256, 0, stream>>>(w_qkv, wqkvb, n4);
    n4 = (CC * CC) / 4;
    cvt_bf16x4<<<(n4 + 255) / 256, 256, 0, stream>>>(w_out, woutb, n4);
  }

  // 1) QKV projection: M=4096 x O=3072, block 128x64 tiles
  qkv_gemm<<<dim3(3 * CC / 64, NB * LL / 128), 256, 0, stream>>>(
      xb, wqkvb, Qb, Kb, Vt);

  // 2) causal flash attention: 4 waves/block, 64 rows/block
  flash_attn<<<dim3(LL / 64, NB * HH), 128, 0, stream>>>(Qb, Kb, Vt, Yb);

  // 3) output projection: M=4096 x O=1024
  out_gemm<<<dim3(CC / 64, NB * LL / 128), 256, 0, stream>>>(Yb, woutb, out);
}
